// GDELayer_39367670235153
// MI455X (gfx1250) — matrix operand order repaired and checked by reading, not compiled, not run on hardware
//
#include <hip/hip_runtime.h>

#define DFEAT 128

typedef __attribute__((ext_vector_type(16))) _Float16 v16h;
typedef __attribute__((ext_vector_type(8)))  float    v8f;

// ---------------------------------------------------------------- utilities
__global__ void k_zero(float* __restrict__ p, long n) {
  long i = (long)blockIdx.x * blockDim.x + threadIdx.x;
  if (i < n) p[i] = 0.0f;
}

__global__ void k_deg(const int* __restrict__ idx, float* __restrict__ deg, int E) {
  int i = blockIdx.x * blockDim.x + threadIdx.x;
  if (i < E) unsafeAtomicAdd(&deg[idx[i]], 1.0f);
}

__global__ void k_inv(float* __restrict__ d, int n) {
  int i = blockIdx.x * blockDim.x + threadIdx.x;
  if (i < n) d[i] = rsqrtf(fmaxf(d[i], 1.0f));
}

// Pack W (DxD f32, row-major, y = x @ W so B-matrix is W[k][col]) into
// WMMA 16-bit B-fragment order: index = (((nt*4 + ks)*32 + lane)*16 + elem).
// Per-lane: col = nt*16 + (lane&15); half e = K = ks*32 + 8*(lane>>4) + (e&7) + 16*(e>>3).
__global__ void k_packW(const float* __restrict__ W, _Float16* __restrict__ Wp) {
  int t = blockIdx.x * blockDim.x + threadIdx.x;
  if (t >= DFEAT * DFEAT) return;
  int e    = t & 15;
  int lane = (t >> 4) & 31;
  int ks   = (t >> 9) & 3;
  int nt   = (t >> 11) & 7;
  int j = e >> 1, odd = e & 1;
  int hh = lane >> 4, lm = lane & 15;
  int k   = ks * 32 + 16 * (j >> 2) + 8 * hh + 2 * (j & 3) + odd;
  int col = nt * 16 + lm;
  Wp[t] = (_Float16)W[k * DFEAT + col];
}

// Per-edge gather (scaled by src out-degree norm) + scatter-add into agg.
// One thread handles a float4 chunk (32 threads per edge).
__global__ void k_scatter(const float* __restrict__ x, const int* __restrict__ src,
                          const int* __restrict__ dst, const float* __restrict__ invout,
                          float* __restrict__ agg, int E) {
  int tid = blockIdx.x * blockDim.x + threadIdx.x;
  int e = tid >> 5;
  if (e >= E) return;
  int c = (tid & 31) << 2;
  int s = src[e], d = dst[e];
  float sc = invout[s];
  float4 v = *(const float4*)(x + (size_t)s * DFEAT + c);
  float* o = agg + (size_t)d * DFEAT + c;
  unsafeAtomicAdd(o + 0, v.x * sc);
  unsafeAtomicAdd(o + 1, v.y * sc);
  unsafeAtomicAdd(o + 2, v.z * sc);
  unsafeAtomicAdd(o + 3, v.w * sc);
}

// out[row][:] = act( (agg[row][:] * invin[row]) @ W + bias )
// One wave32 computes a 16x128 tile: 8 N-tiles x 4 K-steps of v_wmma_f32_16x16x32_f16.
__global__ __launch_bounds__(256) void k_gemm(
    const float* __restrict__ A, const float* __restrict__ invin,
    const _Float16* __restrict__ Wp, const float* __restrict__ bias,
    float* __restrict__ out, int N, int ntiles, int relu) {
  int wave = threadIdx.x >> 5;
  int lane = threadIdx.x & 31;
  int tile = blockIdx.x * 8 + wave;
  if (tile >= ntiles) return;

  int m0 = tile << 4;
  int lm = lane & 15, hh = lane >> 4;
  int row = m0 + lm;
  int rowc = row < N ? row : (N - 1);
  const float* arow = A + (size_t)rowc * DFEAT;
  float s = (row < N) ? invin[rowc] : 0.0f;

  v8f acc[8];
#pragma unroll
  for (int i = 0; i < 8; ++i) acc[i] = (v8f){0.f,0.f,0.f,0.f,0.f,0.f,0.f,0.f};

#pragma unroll
  for (int ks = 0; ks < 4; ++ks) {
    int kbase = ks * 32;
    // A fragment (16-bit A 16x32 layout): halves 0..7 = K [kbase+8*hh .. +7],
    // halves 8..15 = K [kbase+16+8*hh .. +7]  (contiguous float4 x2 each)
    int k0 = kbase + 8 * hh;
    float4 q0 = *(const float4*)(arow + k0);
    float4 q1 = *(const float4*)(arow + k0 + 4);
    int k1 = kbase + 16 + 8 * hh;
    float4 q2 = *(const float4*)(arow + k1);
    float4 q3 = *(const float4*)(arow + k1 + 4);
    v16h a;
    a[0]  = (_Float16)(q0.x * s); a[1]  = (_Float16)(q0.y * s);
    a[2]  = (_Float16)(q0.z * s); a[3]  = (_Float16)(q0.w * s);
    a[4]  = (_Float16)(q1.x * s); a[5]  = (_Float16)(q1.y * s);
    a[6]  = (_Float16)(q1.z * s); a[7]  = (_Float16)(q1.w * s);
    a[8]  = (_Float16)(q2.x * s); a[9]  = (_Float16)(q2.y * s);
    a[10] = (_Float16)(q2.z * s); a[11] = (_Float16)(q2.w * s);
    a[12] = (_Float16)(q3.x * s); a[13] = (_Float16)(q3.y * s);
    a[14] = (_Float16)(q3.z * s); a[15] = (_Float16)(q3.w * s);

#pragma unroll
    for (int nt = 0; nt < 8; ++nt) {
      const v16h b = *(const v16h*)(Wp + (((size_t)(nt * 4 + ks) * 32 + lane) << 4));
      acc[nt] = __builtin_amdgcn_wmma_f32_16x16x32_f16(
          false, a, false, b, (short)0, acc[nt], false, false);
    }
  }

#pragma unroll
  for (int nt = 0; nt < 8; ++nt) {
    int col = nt * 16 + lm;
    float bv = bias[col];
#pragma unroll
    for (int r = 0; r < 8; ++r) {
      int orow = m0 + r + 8 * hh;
      if (orow < N) {
        float v = acc[nt][r] + bv;
        if (relu) v = fmaxf(v, 0.0f);
        out[(size_t)orow * DFEAT + col] = v;
      }
    }
  }
}

// ---------------------------------------------------------------- launcher
extern "C" void kernel_launch(void* const* d_in, const int* in_sizes, int n_in,
                              void* d_out, int out_size, void* d_ws, size_t ws_size,
                              hipStream_t stream) {
  (void)n_in; (void)ws_size;
  const float* h   = (const float*)d_in[1];
  const int*   src = (const int*)d_in[2];
  const int*   dst = (const int*)d_in[3];
  const float* W1  = (const float*)d_in[4];
  const float* b1  = (const float*)d_in[5];
  const float* W2  = (const float*)d_in[6];
  const float* b2  = (const float*)d_in[7];
  float* out = (float*)d_out;

  const int N = in_sizes[1] / DFEAT;
  const int E = in_sizes[2];
  const long aggElems = (long)N * DFEAT;

  // workspace layout (all 32B-aligned for N*DFEAT multiples of 8)
  float* agg     = (float*)d_ws;
  float* inv_out = agg + aggElems;      // deg_out -> inv_sqrt_out
  float* inv_in  = inv_out + N;         // deg_in  -> inv_sqrt_in
  _Float16* W1p  = (_Float16*)(inv_in + N);
  _Float16* W2p  = W1p + DFEAT * DFEAT;

  const int B = 256;
  const int ntiles = (N + 15) / 16;
  const int gemmBlocks = (ntiles + 7) / 8;
  const long zeroAll = aggElems + 2L * N;
  const int scatterBlocks = (int)(((long)E * 32 + B - 1) / B);

  // 1) zero agg + both degree arrays
  k_zero<<<(int)((zeroAll + B - 1) / B), B, 0, stream>>>(agg, zeroAll);
  // 2) degrees
  k_deg<<<(E + B - 1) / B, B, 0, stream>>>(src, inv_out, E);
  k_deg<<<(E + B - 1) / B, B, 0, stream>>>(dst, inv_in, E);
  // 3) clip + rsqrt (both arrays contiguous)
  k_inv<<<(2 * N + B - 1) / B, B, 0, stream>>>(inv_out, 2 * N);
  // 4) pack weights to f16 B-fragment order
  k_packW<<<(DFEAT * DFEAT + B - 1) / B, B, 0, stream>>>(W1, W1p);
  k_packW<<<(DFEAT * DFEAT + B - 1) / B, B, 0, stream>>>(W2, W2p);

  // ---- layer 1: h -> d_out (ReLU)
  k_scatter<<<scatterBlocks, B, 0, stream>>>(h, src, dst, inv_out, agg, E);
  k_gemm<<<gemmBlocks, B, 0, stream>>>(agg, inv_in, W1p, b1, out, N, ntiles, 1);

  // ---- layer 2: d_out -> d_out (no activation)
  k_zero<<<(int)((aggElems + B - 1) / B), B, 0, stream>>>(agg, aggElems);
  k_scatter<<<scatterBlocks, B, 0, stream>>>(out, src, dst, inv_out, agg, E);
  k_gemm<<<gemmBlocks, B, 0, stream>>>(agg, inv_in, W2p, b2, out, N, ntiles, 0);
}
